// PointNetShapeServoKp2_32014686224760
// MI455X (gfx1250) — compile-verified, matrix operand order checked by reading, not run on hardware
//
#include <hip/hip_runtime.h>
#include <hip/hip_bf16.h>

// ---------------------------------------------------------------------------
// PointNet/PointConv shape-servo network for MI455X (gfx1250, wave32, WMMA).
// Dense contractions run as f16 WMMA GEMMs: A and W are pre-packed to
// zero-padded f16 (K padded to a multiple of 32), so the inner loop is
// 4x global_load_b128 + v_wmma_f32_16x16x32_f16, branch-free, unrolled x4.
// ---------------------------------------------------------------------------

typedef __attribute__((ext_vector_type(16))) _Float16 v16h;
typedef __attribute__((ext_vector_type(8)))  _Float16 v8h;
typedef __attribute__((ext_vector_type(8)))  float    v8f;

#define BN_SCALE_F 0.9999950000374997f   // 1/sqrt(1 + 1e-5)

__device__ __forceinline__ float bn_ev(float x, float g, float b) {
  return x * (BN_SCALE_F * g) + b;
}

// ---------------------------------------------------------------------------
// 1) extract: x (B,6,N) -> xyz (B,N,3), pts (B,N,6)
// ---------------------------------------------------------------------------
__global__ void extract_kernel(const float* __restrict__ x,
                               float* __restrict__ xyz,
                               float* __restrict__ pts, int N) {
  int b = blockIdx.y;
  int i = blockIdx.x * blockDim.x + threadIdx.x;
  if (i >= N) return;
  for (int c = 0; c < 6; ++c) {
    float v = x[((size_t)b * 6 + c) * N + i];
    pts[((size_t)b * N + i) * 6 + c] = v;
    if (c < 3) xyz[((size_t)b * N + i) * 3 + c] = v;
  }
}

// ---------------------------------------------------------------------------
// 2) density: inv_d[b,i] = 1 / mean_j( exp(-||xi-xj||^2/(2 bw^2)) / (2.5 bw) )
// ---------------------------------------------------------------------------
__global__ void density_kernel(const float* __restrict__ xyz,
                               float* __restrict__ inv_d, int N, float bw) {
  int b = blockIdx.y;
  int i = blockIdx.x * blockDim.x + threadIdx.x;
  const float* base = xyz + (size_t)b * N * 3;
  float xi = 0.f, yi = 0.f, zi = 0.f;
  if (i < N) { xi = base[i*3]; yi = base[i*3+1]; zi = base[i*3+2]; }
  __shared__ float tile[256 * 3];
  float sum = 0.f;
  float inv2 = 1.0f / (2.0f * bw * bw);
  for (int j0 = 0; j0 < N; j0 += 256) {
    int jn = N - j0; if (jn > 256) jn = 256;
    __syncthreads();
    for (int t = threadIdx.x; t < jn * 3; t += blockDim.x) tile[t] = base[j0*3 + t];
    __syncthreads();
    for (int j = 0; j < jn; ++j) {
      float dx = xi - tile[j*3+0];
      float dy = yi - tile[j*3+1];
      float dz = zi - tile[j*3+2];
      float d2 = dx*dx + dy*dy + dz*dz;
      sum += __expf(-d2 * inv2);
    }
  }
  if (i < N) {
    float dens = (sum / (2.5f * bw)) / (float)N;
    inv_d[(size_t)b * N + i] = 1.0f / dens;
  }
}

// ---------------------------------------------------------------------------
// 3) farthest point sampling: one 256-thread block per batch
// ---------------------------------------------------------------------------
__global__ void fps_kernel(const float* __restrict__ xyz, int* __restrict__ fps_idx,
                           int N, int npoint) {
  int b = blockIdx.x;
  const float* base = xyz + (size_t)b * N * 3;
  __shared__ float dist[2048];
  __shared__ float redv[256];
  __shared__ int   redi[256];
  __shared__ int   s_far;
  int tid = threadIdx.x;
  for (int i = tid; i < N; i += blockDim.x) dist[i] = 1e10f;
  if (tid == 0) s_far = 0;
  __syncthreads();
  for (int it = 0; it < npoint; ++it) {
    int far = s_far;
    if (tid == 0) fps_idx[b * npoint + it] = far;
    float cx = base[far*3], cy = base[far*3+1], cz = base[far*3+2];
    float best = -1.f; int besti = 0;
    for (int i = tid; i < N; i += blockDim.x) {
      float dx = base[i*3]-cx, dy = base[i*3+1]-cy, dz = base[i*3+2]-cz;
      float d = dx*dx + dy*dy + dz*dz;
      float nd = dist[i]; if (d < nd) nd = d;
      dist[i] = nd;
      if (nd > best) { best = nd; besti = i; }
    }
    redv[tid] = best; redi[tid] = besti;
    __syncthreads();
    for (int off = 128; off > 0; off >>= 1) {
      if (tid < off) {
        bool take = (redv[tid+off] > redv[tid]) ||
                    (redv[tid+off] == redv[tid] && redi[tid+off] < redi[tid]);
        if (take) { redv[tid] = redv[tid+off]; redi[tid] = redi[tid+off]; }
      }
      __syncthreads();
    }
    if (tid == 0) s_far = redi[0];
    __syncthreads();
  }
}

// ---------------------------------------------------------------------------
// 4) kNN + grouping: block per (b,s).
// ---------------------------------------------------------------------------
__global__ void knn_group_kernel(const float* __restrict__ xyz,
                                 const float* __restrict__ pts,
                                 const float* __restrict__ invd,
                                 const int* __restrict__ fps_idx,
                                 float* __restrict__ new_xyz,
                                 float* __restrict__ gnp,
                                 float* __restrict__ ginvd,
                                 int N, int S, int K, int Cpts) {
  int s = blockIdx.x, b = blockIdx.y;
  const float* xb = xyz + (size_t)b * N * 3;
  __shared__ float dist[2048];
  __shared__ float redv[256];
  __shared__ int   redi[256];
  __shared__ int   s_sel;
  int tid = threadIdx.x;
  int ci = fps_idx[b * S + s];
  float cx = xb[ci*3], cy = xb[ci*3+1], cz = xb[ci*3+2];
  if (tid == 0) {
    float* nx = new_xyz + ((size_t)b * S + s) * 3;
    nx[0] = cx; nx[1] = cy; nx[2] = cz;
  }
  for (int i = tid; i < N; i += blockDim.x) {
    float dx = xb[i*3]-cx, dy = xb[i*3+1]-cy, dz = xb[i*3+2]-cz;
    dist[i] = dx*dx + dy*dy + dz*dz;
  }
  __syncthreads();
  int C = 3 + Cpts;
  for (int k = 0; k < K; ++k) {
    float best = 3.4e38f; int besti = N;
    for (int i = tid; i < N; i += blockDim.x) {
      float d = dist[i];
      if (d < best) { best = d; besti = i; }
    }
    redv[tid] = best; redi[tid] = besti;
    __syncthreads();
    for (int off = 128; off > 0; off >>= 1) {
      if (tid < off) {
        bool take = (redv[tid+off] < redv[tid]) ||
                    (redv[tid+off] == redv[tid] && redi[tid+off] < redi[tid]);
        if (take) { redv[tid] = redv[tid+off]; redi[tid] = redi[tid+off]; }
      }
      __syncthreads();
    }
    if (tid == 0) s_sel = (redi[0] < N) ? redi[0] : 0;
    __syncthreads();
    int j = s_sel;
    float* nprow = gnp + (((size_t)(b * S + s)) * K + k) * C;
    if (tid == 0) {
      nprow[0] = xb[j*3+0] - cx;
      nprow[1] = xb[j*3+1] - cy;
      nprow[2] = xb[j*3+2] - cz;
      ginvd[((size_t)(b * S + s)) * K + k] = invd[(size_t)b * N + j];
      dist[j] = 3.0e38f;   // mask for next selection round
    }
    for (int c = tid; c < Cpts; c += blockDim.x)
      nprow[3 + c] = pts[((size_t)b * N + j) * Cpts + c];
    __syncthreads();
  }
}

// ---------------------------------------------------------------------------
// 5a) pack: f32 (rows, K) -> f16 (rows, Kp) zero-padded
// ---------------------------------------------------------------------------
__global__ void pack_f16_kernel(const float* __restrict__ src,
                                _Float16* __restrict__ dst,
                                int rows, int K, int Kp) {
  size_t total = (size_t)rows * Kp;
  size_t stride = (size_t)gridDim.x * blockDim.x;
  for (size_t t = (size_t)blockIdx.x * blockDim.x + threadIdx.x; t < total; t += stride) {
    int r = (int)(t / Kp);
    int c = (int)(t % Kp);
    dst[t] = (c < K) ? (_Float16)src[(size_t)r * K + c] : (_Float16)0.f;
  }
}

// ---------------------------------------------------------------------------
// 5b) WMMA GEMM: out(M,N) = epilogue( A(M,Kp) * W(N,Kp)^T ), f16 in, f32 acc.
//     M, N multiples of 16; Kp multiple of 32.  One wave per 16x16 tile.
//     Lane layout (ISA 7.12.2): 16-bit A 16x32 -> per lane two contiguous
//     8-half (16B) chunks at k0 + half*8 and k0 + 16 + half*8.
// ---------------------------------------------------------------------------
__global__ void wmma_gemm_kernel(const _Float16* __restrict__ A,
                                 const _Float16* __restrict__ W,
                                 const float* __restrict__ bias,
                                 const float* __restrict__ gamma,
                                 const float* __restrict__ beta,
                                 float* __restrict__ out,
                                 int M, int N, int Kp, int relu_bn) {
  int wave = threadIdx.x >> 5;
  int lane = threadIdx.x & 31;
  int tiles_n = N >> 4;
  int tiles_m = M >> 4;
  int tile_id = blockIdx.x * 8 + wave;
  if (tile_id >= tiles_m * tiles_n) return;   // wave-uniform: EXEC stays full
  int tile_m = tile_id / tiles_n;
  int tile_n = tile_id % tiles_n;
  int half = lane >> 4;
  int mr   = lane & 15;
  const _Float16* Ap = A + (size_t)(tile_m * 16 + mr) * Kp + half * 8;
  const _Float16* Wp = W + (size_t)(tile_n * 16 + mr) * Kp + half * 8;
  v8f acc = {};
#pragma unroll 4
  for (int k0 = 0; k0 < Kp; k0 += 32) {
    union { v16h v; v8h h[2]; } a, b;
    a.h[0] = *(const v8h*)(Ap + k0);
    a.h[1] = *(const v8h*)(Ap + k0 + 16);
    b.h[0] = *(const v8h*)(Wp + k0);
    b.h[1] = *(const v8h*)(Wp + k0 + 16);
    acc = __builtin_amdgcn_wmma_f32_16x16x32_f16(
        /*neg_a=*/false, a.v, /*neg_b=*/false, b.v,
        /*c_mod=*/(short)0, acc, /*reuse_a=*/false, /*reuse_b=*/false);
  }
  int n = tile_n * 16 + mr;
  float bi = bias[n];
  float ga = relu_bn ? gamma[n] : 0.f;
  float be = relu_bn ? beta[n]  : 0.f;
#pragma unroll
  for (int r = 0; r < 8; ++r) {
    int m = tile_m * 16 + r + half * 8;     // C/D layout: VGPR r; halves M / M+8
    float v = acc[r] + bi;
    if (relu_bn) v = fmaxf(bn_ev(v, ga, be), 0.f);
    out[(size_t)m * N + n] = v;
  }
}

// ---------------------------------------------------------------------------
// 6) weight-net MLP 3->8->8->16 (+ density scale).  blockDim == K (8/16/64).
// ---------------------------------------------------------------------------
__global__ void wnet_kernel(const float* __restrict__ gxyz, int kstride,
                            const float* __restrict__ ginvd,
                            const float* __restrict__ w0, const float* __restrict__ b0,
                            const float* __restrict__ g0, const float* __restrict__ be0,
                            const float* __restrict__ w1, const float* __restrict__ b1,
                            const float* __restrict__ g1, const float* __restrict__ be1,
                            const float* __restrict__ w2, const float* __restrict__ b2,
                            const float* __restrict__ g2, const float* __restrict__ be2,
                            float* __restrict__ wout, int K) {
  int bs = blockIdx.x;
  int k = threadIdx.x;
  __shared__ float red[64];
  const float* g = gxyz + ((size_t)bs * K + k) * kstride;
  float x0 = g[0], x1 = g[1], x2 = g[2];
  float h0[8];
#pragma unroll
  for (int o = 0; o < 8; ++o) {
    float v = w0[o*3+0]*x0 + w0[o*3+1]*x1 + w0[o*3+2]*x2 + b0[o];
    h0[o] = fmaxf(bn_ev(v, g0[o], be0[o]), 0.f);
  }
  float h1[8];
#pragma unroll
  for (int o = 0; o < 8; ++o) {
    float v = b1[o];
#pragma unroll
    for (int i = 0; i < 8; ++i) v += w1[o*8+i] * h0[i];
    h1[o] = fmaxf(bn_ev(v, g1[o], be1[o]), 0.f);
  }
  float h2[16];
#pragma unroll
  for (int o = 0; o < 16; ++o) {
    float v = b2[o];
#pragma unroll
    for (int i = 0; i < 8; ++i) v += w2[o*8+i] * h1[i];
    h2[o] = fmaxf(bn_ev(v, g2[o], be2[o]), 0.f);
  }
  float id = ginvd[(size_t)bs * K + k];
  red[k] = id;
  __syncthreads();
  for (int off = blockDim.x >> 1; off > 0; off >>= 1) {
    if (k < off) red[k] = fmaxf(red[k], red[k + off]);
    __syncthreads();
  }
  float scale = id / red[0];
#pragma unroll
  for (int w = 0; w < 16; ++w)
    wout[((size_t)bs * K + k) * 16 + w] = h2[w] * scale;
}

// ---------------------------------------------------------------------------
// 7) aggregation: agg[bs, o*16+w] = sum_k h[bs,k,o] * wnet[bs,k,w]
// ---------------------------------------------------------------------------
__global__ void agg_kernel(const float* __restrict__ h,
                           const float* __restrict__ wn,
                           float* __restrict__ agg, int K, int O) {
  int bs = blockIdx.x;
  const float* hb = h  + (size_t)bs * K * O;
  const float* wb = wn + (size_t)bs * K * 16;
  __shared__ float ws_[64 * 16];
  for (int t = threadIdx.x; t < K * 16; t += blockDim.x) ws_[t] = wb[t];
  __syncthreads();
  for (int t = threadIdx.x; t < O * 16; t += blockDim.x) {
    int o = t >> 4, wi = t & 15;
    float s = 0.f;
    for (int k = 0; k < K; ++k) s += hb[k * O + o] * ws_[k * 16 + wi];
    agg[(size_t)bs * O * 16 + t] = s;
  }
}

// ---------------------------------------------------------------------------
// 8) sa3 group_all concat: np (B,Np,3+Cp) = [xyz | pts]
// ---------------------------------------------------------------------------
__global__ void concat_kernel(const float* __restrict__ xyz,
                              const float* __restrict__ pts,
                              float* __restrict__ np, int Np, int Cp) {
  int b = blockIdx.x;
  int C = 3 + Cp;
  for (int t = threadIdx.x; t < Np * C; t += blockDim.x) {
    int i = t / C, c = t % C;
    float v = (c < 3) ? xyz[((size_t)b * Np + i) * 3 + c]
                      : pts[((size_t)b * Np + i) * Cp + (c - 3)];
    np[((size_t)b * Np + i) * C + c] = v;
  }
}

// ---------------------------------------------------------------------------
// 9) head helpers
// ---------------------------------------------------------------------------
__global__ void sub_kernel(const float* __restrict__ a, const float* __restrict__ b,
                           float* __restrict__ o, int n) {
  int i = blockIdx.x * blockDim.x + threadIdx.x;
  if (i < n) o[i] = a[i] - b[i];
}

__global__ void fc_gn_kernel(const float* __restrict__ x, const float* __restrict__ W,
                             const float* __restrict__ bias,
                             const float* __restrict__ gg, const float* __restrict__ gb,
                             float* __restrict__ y, int K, int N, int do_gn) {
  int b = blockIdx.x;
  int t = threadIdx.x;
  __shared__ float red[128];
  __shared__ float s_mean, s_var;
  float v = 0.f;
  if (t < N) {
    const float* w  = W + (size_t)t * K;
    const float* xb = x + (size_t)b * K;
    float acc = bias[t];
    for (int i = 0; i < K; ++i) acc += w[i] * xb[i];
    v = acc;
  }
  if (do_gn) {
    red[t] = (t < N) ? v : 0.f;
    __syncthreads();
    for (int o = blockDim.x >> 1; o > 0; o >>= 1) { if (t < o) red[t] += red[t+o]; __syncthreads(); }
    if (t == 0) s_mean = red[0] / (float)N;
    __syncthreads();
    float m = s_mean;
    float d = (t < N) ? (v - m) : 0.f;
    red[t] = d * d;
    __syncthreads();
    for (int o = blockDim.x >> 1; o > 0; o >>= 1) { if (t < o) red[t] += red[t+o]; __syncthreads(); }
    if (t == 0) s_var = red[0] / (float)N;
    __syncthreads();
    if (t < N) {
      v = (v - m) * rsqrtf(s_var + 1e-5f) * gg[t] + gb[t];
      v = fmaxf(v, 0.f);
    }
  }
  if (t < N) y[(size_t)b * N + t] = v;
}

// ---------------------------------------------------------------------------
// Host-side orchestration
// ---------------------------------------------------------------------------
static inline int ceil_div(int a, int b) { return (a + b - 1) / b; }

static void launch_gemm(const float* A, const float* W, const float* bias,
                        const float* g, const float* be, float* out,
                        _Float16* Ah, _Float16* Wh,
                        int M, int N, int K, int relu_bn, hipStream_t st) {
  int Kp = ceil_div(K, 32) * 32;
  {
    size_t ta = (size_t)M * Kp;
    int blk = (int)((ta + 255) / 256); if (blk > 2048) blk = 2048;
    pack_f16_kernel<<<blk, 256, 0, st>>>(A, Ah, M, K, Kp);
    size_t tw = (size_t)N * Kp;
    blk = (int)((tw + 255) / 256); if (blk > 2048) blk = 2048;
    pack_f16_kernel<<<blk, 256, 0, st>>>(W, Wh, N, K, Kp);
  }
  int tiles = (M >> 4) * (N >> 4);
  int blocks = ceil_div(tiles, 8);
  wmma_gemm_kernel<<<blocks, 256, 0, st>>>(Ah, Wh, bias, g, be, out, M, N, Kp, relu_bn);
}

struct SAP {
  const float *conv_w, *conv_b, *conv_g, *conv_be;
  const float *w0, *w1, *w2, *b0, *b1, *b2, *g0, *g1, *g2, *be0, *be1, *be2;
  const float *lin_w, *lin_b, *bnl_g, *bnl_b;
};

static SAP sa_from(void* const* d_in, int base) {
  SAP p;
  p.conv_w  = (const float*)d_in[base + 0];
  p.conv_b  = (const float*)d_in[base + 1];
  p.conv_g  = (const float*)d_in[base + 2];
  p.conv_be = (const float*)d_in[base + 3];
  p.w0  = (const float*)d_in[base + 4];
  p.w1  = (const float*)d_in[base + 5];
  p.w2  = (const float*)d_in[base + 6];
  p.b0  = (const float*)d_in[base + 7];
  p.b1  = (const float*)d_in[base + 8];
  p.b2  = (const float*)d_in[base + 9];
  p.g0  = (const float*)d_in[base + 10];
  p.g1  = (const float*)d_in[base + 11];
  p.g2  = (const float*)d_in[base + 12];
  p.be0 = (const float*)d_in[base + 13];
  p.be1 = (const float*)d_in[base + 14];
  p.be2 = (const float*)d_in[base + 15];
  p.lin_w = (const float*)d_in[base + 16];
  p.lin_b = (const float*)d_in[base + 17];
  p.bnl_g = (const float*)d_in[base + 18];
  p.bnl_b = (const float*)d_in[base + 19];
  return p;
}

static void run_encode(const float* x6N, const SAP& s1, const SAP& s2, const SAP& s3,
                       float* ws, float* l3, hipStream_t st) {
  const int B = 16;
  size_t off = 0;
  auto alloc = [&](size_t n) { float* p = ws + off; off += n; return p; };
  float* xyzA  = alloc((size_t)B * 2048 * 3);
  float* ptsA  = alloc((size_t)B * 2048 * 6);
  float* invd1 = alloc((size_t)B * 2048);
  float* nx1   = alloc((size_t)B * 128 * 3);
  float* gnp1  = alloc((size_t)B * 128 * 8 * 9);
  float* gid1  = alloc((size_t)B * 128 * 8);
  float* h1    = alloc((size_t)B * 128 * 8 * 64);
  float* wn1   = alloc((size_t)B * 128 * 8 * 16);
  float* agg1  = alloc((size_t)B * 128 * 1024);
  float* l1p   = alloc((size_t)B * 128 * 64);
  float* invd2 = alloc((size_t)B * 128);
  float* nx2   = alloc((size_t)B * 64 * 3);
  float* gnp2  = alloc((size_t)B * 64 * 16 * 67);
  float* gid2  = alloc((size_t)B * 64 * 16);
  float* h2    = alloc((size_t)B * 64 * 16 * 128);
  float* wn2   = alloc((size_t)B * 64 * 16 * 16);
  float* agg2  = alloc((size_t)B * 64 * 2048);
  float* l2p   = alloc((size_t)B * 64 * 128);
  float* invd3 = alloc((size_t)B * 64);
  float* np3   = alloc((size_t)B * 64 * 131);
  float* h3    = alloc((size_t)B * 64 * 256);
  float* wn3   = alloc((size_t)B * 64 * 16);
  float* agg3  = alloc((size_t)B * 4096);
  int* fps1 = (int*)alloc((size_t)B * 128);
  int* fps2 = (int*)alloc((size_t)B * 64);
  // f16 packing staging (sized for the largest GEMM operands)
  _Float16* Ah = (_Float16*)alloc((size_t)2 * 1024 * 1024 / 2 * 2 / 2 + 1048576); // 4.2M halves
  _Float16* Wh = (_Float16*)alloc(1048576 / 2 + 524288);                          // 2.1M halves

  // ---- SA1: N=2048, S=128, K=8, C=9, O=64, bw=0.1 ----
  extract_kernel<<<dim3(8, B), 256, 0, st>>>(x6N, xyzA, ptsA, 2048);
  density_kernel<<<dim3(8, B), 256, 0, st>>>(xyzA, invd1, 2048, 0.1f);
  fps_kernel<<<B, 256, 0, st>>>(xyzA, fps1, 2048, 128);
  knn_group_kernel<<<dim3(128, B), 256, 0, st>>>(xyzA, ptsA, invd1, fps1,
                                                 nx1, gnp1, gid1, 2048, 128, 8, 6);
  launch_gemm(gnp1, s1.conv_w, s1.conv_b, s1.conv_g, s1.conv_be, h1,
              Ah, Wh, B * 128 * 8, 64, 9, 1, st);
  wnet_kernel<<<B * 128, 8, 0, st>>>(gnp1, 9, gid1,
      s1.w0, s1.b0, s1.g0, s1.be0, s1.w1, s1.b1, s1.g1, s1.be1,
      s1.w2, s1.b2, s1.g2, s1.be2, wn1, 8);
  agg_kernel<<<B * 128, 256, 0, st>>>(h1, wn1, agg1, 8, 64);
  launch_gemm(agg1, s1.lin_w, s1.lin_b, s1.bnl_g, s1.bnl_b, l1p,
              Ah, Wh, B * 128, 64, 1024, 1, st);

  // ---- SA2: N=128, S=64, K=16, C=67, O=128, bw=0.2 ----
  density_kernel<<<dim3(1, B), 256, 0, st>>>(nx1, invd2, 128, 0.2f);
  fps_kernel<<<B, 256, 0, st>>>(nx1, fps2, 128, 64);
  knn_group_kernel<<<dim3(64, B), 256, 0, st>>>(nx1, l1p, invd2, fps2,
                                                nx2, gnp2, gid2, 128, 64, 16, 64);
  launch_gemm(gnp2, s2.conv_w, s2.conv_b, s2.conv_g, s2.conv_be, h2,
              Ah, Wh, B * 64 * 16, 128, 67, 1, st);
  wnet_kernel<<<B * 64, 16, 0, st>>>(gnp2, 67, gid2,
      s2.w0, s2.b0, s2.g0, s2.be0, s2.w1, s2.b1, s2.g1, s2.be1,
      s2.w2, s2.b2, s2.g2, s2.be2, wn2, 16);
  agg_kernel<<<B * 64, 256, 0, st>>>(h2, wn2, agg2, 16, 128);
  launch_gemm(agg2, s2.lin_w, s2.lin_b, s2.bnl_g, s2.bnl_b, l2p,
              Ah, Wh, B * 64, 128, 2048, 1, st);

  // ---- SA3 (group_all): Np=64, C=131, O=256, bw=0.4 ----
  density_kernel<<<dim3(1, B), 256, 0, st>>>(nx2, invd3, 64, 0.4f);
  concat_kernel<<<B, 256, 0, st>>>(nx2, l2p, np3, 64, 128);
  launch_gemm(np3, s3.conv_w, s3.conv_b, s3.conv_g, s3.conv_be, h3,
              Ah, Wh, B * 64, 256, 131, 1, st);
  wnet_kernel<<<B, 64, 0, st>>>(nx2, 3, invd3,
      s3.w0, s3.b0, s3.g0, s3.be0, s3.w1, s3.b1, s3.g1, s3.be1,
      s3.w2, s3.b2, s3.g2, s3.be2, wn3, 64);
  agg_kernel<<<B, 256, 0, st>>>(h3, wn3, agg3, 64, 256);
  launch_gemm(agg3, s3.lin_w, s3.lin_b, s3.bnl_g, s3.bnl_b, l3,
              Ah, Wh, B, 256, 4096, 1, st);
}

extern "C" void kernel_launch(void* const* d_in, const int* in_sizes, int n_in,
                              void* d_out, int out_size, void* d_ws, size_t ws_size,
                              hipStream_t stream) {
  (void)in_sizes; (void)n_in; (void)out_size; (void)ws_size;
  const int B = 16;
  const float* xyz      = (const float*)d_in[0];
  const float* xyz_goal = (const float*)d_in[1];
  SAP s1 = sa_from(d_in, 2);
  SAP s2 = sa_from(d_in, 22);
  SAP s3 = sa_from(d_in, 42);
  const float* fc1_w = (const float*)d_in[62];
  const float* fc1_b = (const float*)d_in[63];
  const float* gn1_g = (const float*)d_in[64];
  const float* gn1_b = (const float*)d_in[65];
  const float* fc3_w = (const float*)d_in[66];
  const float* fc3_b = (const float*)d_in[67];
  const float* gn3_g = (const float*)d_in[68];
  const float* gn3_b = (const float*)d_in[69];
  const float* fc4_w = (const float*)d_in[70];
  const float* fc4_b = (const float*)d_in[71];
  const float* gn4_g = (const float*)d_in[72];
  const float* gn4_b = (const float*)d_in[73];
  const float* fc5_w = (const float*)d_in[74];
  const float* fc5_b = (const float*)d_in[75];

  float* ws = (float*)d_ws;
  // head region (first 16K floats), encode scratch after
  float* l3x  = ws + 0;
  float* l3g  = ws + 4096;
  float* feat = ws + 8192;
  float* hb1  = ws + 12288;                 // (B,128)
  float* hb3  = hb1 + B * 128;              // (B,64)
  float* hb4  = hb3 + B * 64;               // (B,32)
  float* enc_ws = ws + 16384;

  run_encode(xyz,      s1, s2, s3, enc_ws, l3x, stream);
  run_encode(xyz_goal, s1, s2, s3, enc_ws, l3g, stream);

  sub_kernel<<<16, 256, 0, stream>>>(l3g, l3x, feat, B * 256);
  fc_gn_kernel<<<B, 128, 0, stream>>>(feat, fc1_w, fc1_b, gn1_g, gn1_b, hb1, 256, 128, 1);
  fc_gn_kernel<<<B, 64, 0, stream>>>(hb1, fc3_w, fc3_b, gn3_g, gn3_b, hb3, 128, 64, 1);
  fc_gn_kernel<<<B, 32, 0, stream>>>(hb3, fc4_w, fc4_b, gn4_g, gn4_b, hb4, 64, 32, 1);
  fc_gn_kernel<<<B, 32, 0, stream>>>(hb4, fc5_w, fc5_b, nullptr, nullptr,
                                     (float*)d_out, 32, 3, 0);
}
